// HybridReIDModel_56702158242289
// MI455X (gfx1250) — compile-verified, matrix operand order checked by reading, not run on hardware
//
#include <hip/hip_runtime.h>
#include <cstdint>
#include <cstddef>

typedef __bf16 bf16;
typedef __bf16 v8bf  __attribute__((ext_vector_type(8)));
typedef __bf16 v16bf __attribute__((ext_vector_type(16)));
typedef float  v8f   __attribute__((ext_vector_type(8)));
typedef __attribute__((address_space(3))) const bf16 lds_cbf16;

__device__ __forceinline__ v8bf v8bf_zero() {
  v8bf v;
#pragma unroll
  for (int i = 0; i < 8; ++i) v[i] = (bf16)0.0f;
  return v;
}

__device__ __forceinline__ float wave_sum(float v) {
#pragma unroll
  for (int o = 16; o >= 1; o >>= 1) v += __shfl_xor(v, o, 32);
  return v;
}

// Async copy of 16B from global to LDS (per active lane), tracked by ASYNCcnt.
__device__ __forceinline__ void async_load_b128(const bf16* gp, bf16* lp) {
  unsigned lofs = (unsigned)(uintptr_t)(lds_cbf16*)lp;   // LDS byte offset
  asm volatile("global_load_async_to_lds_b128 %0, %1, off"
               :: "v"(lofs), "v"(gp) : "memory");
}

__device__ __forceinline__ void wait_async_le4() {
#if __has_builtin(__builtin_amdgcn_s_wait_asynccnt)
  __builtin_amdgcn_s_wait_asynccnt(4);
#else
  asm volatile("s_wait_asynccnt 0x4" ::: "memory");
#endif
}

__device__ __forceinline__ void wait_async_le0() {
#if __has_builtin(__builtin_amdgcn_s_wait_asynccnt)
  __builtin_amdgcn_s_wait_asynccnt(0);
#else
  asm volatile("s_wait_asynccnt 0x0" ::: "memory");
#endif
}

// ---------------------------------------------------------------------------
// Implicit-GEMM conv, NHWC bf16 activations, [Cout][K] bf16 weights.
// Block = 256 threads = 8 waves arranged 4(M) x 2(N); block tile 64M x 128N.
// K ordered ((ky*kw+kx)*Cin + ci), processed 64 channels per stage (all Cin
// here are multiples of 64) with DOUBLE-BUFFERED LDS tiles:
//   A tile 64x64 bf16  x2 (16KB): global_load_b128 -> regs (prefetched one
//                                 stage ahead, zero-filled halo) -> ds_store
//   B tile 128x64 bf16 x2 (32KB): global_load_async_to_lds_b128, issued one
//                                 stage ahead; ASYNCcnt completes in order so
//                                 s_wait_asynccnt<=4 retires only the current
//                                 stage and leaves the prefetch in flight.
// Each wave: 16M x 64N, 8 x v_wmma_f32_16x16x32_bf16 per stage.
// Epilogue: y = acc*scale[c] + bias[c] (+ residual) (+ ReLU) -> bf16 NHWC.
// ---------------------------------------------------------------------------
__global__ void __launch_bounds__(256)
gemm_conv_wmma(const bf16* __restrict__ A, const bf16* __restrict__ Wp,
               const float* __restrict__ scale, const float* __restrict__ bias,
               const bf16* __restrict__ res, bf16* __restrict__ out,
               int Nb, int H, int W, int Cin, int Cout,
               int kh, int kw, int stride, int pad, int Ho, int Wo, int relu)
{
  __shared__ __align__(16) bf16 As[2][64 * 64];
  __shared__ __align__(16) bf16 Bs[2][128 * 64];

  const int M     = Nb * Ho * Wo;
  const int mBase = blockIdx.y * 64;
  const int nBase = blockIdx.x * 128;
  const int tid   = threadIdx.x;
  const int lane  = tid & 31;
  const int wid   = tid >> 5;
  const int waveM = wid & 3;               // 0..3
  const int waveN = wid >> 2;              // 0..1
  const int l16   = lane & 15;
  const int koff  = (lane >> 4) << 3;      // 0 or 8 (16-bit A/B VGPR layout)

  // ---- A staging map: thread -> (row, 32B chunk) : 64 rows x 64 ci ----
  const int arow = tid >> 2;               // 0..63
  const int aco  = (tid & 3) * 16;         // 0,16,32,48
  const int am   = mBase + arow;
  const bool amv = (am < M);
  int nb = 0, ho = 0, wo = 0;
  if (amv) { wo = am % Wo; int t = am / Wo; ho = t % Ho; nb = t / Ho; }
  const int hi0 = ho * stride - pad;
  const int wi0 = wo * stride - pad;

  // ---- B staging map: thread -> (col, 64B half-row) : 128 cols x 64 ci ----
  // bcol spans one 16-col group per wave and Cout % 16 == 0, so bv is
  // wave-uniform: per-wave ASYNCcnt is exactly 4 per issued stage.
  const int Ktot  = kh * kw * Cin;
  const int bcol  = tid >> 1;              // 0..127
  const int bko   = (tid & 1) * 32;        // 0 or 32
  const bool bv   = (nBase + bcol) < Cout;
  const bf16* bgp = Wp + (size_t)(nBase + bcol) * Ktot + bko;

  // Load a 32B A sliver for stage (ky,kx,cc) into regs (zero outside halo).
  auto load_a = [&](int ky, int kx, int cc, v8bf& a0, v8bf& a1) {
    const int hi = hi0 + ky, wi = wi0 + kx;
    a0 = v8bf_zero(); a1 = a0;
    if (amv && hi >= 0 && hi < H && wi >= 0 && wi < W) {
      const bf16* p = A + (((size_t)nb * H + hi) * W + wi) * Cin + cc + aco;
      a0 = *(const v8bf*)p;
      a1 = *(const v8bf*)(p + 8);
    }
  };

  // Issue this thread's 4 async 16B copies of the B tile for a stage.
  auto issue_b = [&](int ky, int kx, int cc, int buf) {
    if (bv) {
      const bf16* src = bgp + (size_t)(ky * kw + kx) * Cin + cc;
      bf16* dst = &Bs[buf][bcol * 64 + bko];
#pragma unroll
      for (int i = 0; i < 4; ++i)
        async_load_b128(src + i * 8, dst + i * 8);
    }
  };

  v8f acc[4];
#pragma unroll
  for (int j = 0; j < 4; ++j)
#pragma unroll
    for (int i = 0; i < 8; ++i) acc[j][i] = 0.0f;

  const int nstage = kh * kw * (Cin >> 6);

  // ---- pipeline prologue: stage 0 ----
  int ky = 0, kx = 0, cc = 0;
  issue_b(0, 0, 0, 0);
  v8bf a0, a1;
  load_a(0, 0, 0, a0, a1);

  for (int s = 0; s < nstage; ++s) {
    const int cur = s & 1;
    const int nxt = cur ^ 1;
    // stage s+1 indices
    int ky2 = ky, kx2 = kx, cc2 = cc + 64;
    if (cc2 >= Cin) { cc2 = 0; if (++kx2 >= kw) { kx2 = 0; ++ky2; } }
    const bool have_next = (s + 1 < nstage);

    if (have_next) issue_b(ky2, kx2, cc2, nxt);      // B prefetch (async)
    *(v8bf*)&As[cur][arow * 64 + aco]     = a0;      // A stage s -> LDS
    *(v8bf*)&As[cur][arow * 64 + aco + 8] = a1;
    if (have_next) load_a(ky2, kx2, cc2, a0, a1);    // A prefetch -> regs

    if (have_next) wait_async_le4();                  // stage s B done,
    else           wait_async_le0();                  // prefetch still flying
    __syncthreads();

    // ---- two 32-K sub-steps of WMMA ----
#pragma unroll
    for (int ks = 0; ks < 64; ks += 32) {
      const bf16* ap = &As[cur][(waveM * 16 + l16) * 64 + ks + koff];
      v8bf alo = *(const v8bf*)ap;
      v8bf ahi = *(const v8bf*)(ap + 16);
      v16bf afrag = __builtin_shufflevector(alo, ahi,
          0,1,2,3,4,5,6,7,8,9,10,11,12,13,14,15);
#pragma unroll
      for (int j = 0; j < 4; ++j) {
        const bf16* bp = &Bs[cur][(waveN * 64 + j * 16 + l16) * 64 + ks + koff];
        v8bf blo = *(const v8bf*)bp;
        v8bf bhi = *(const v8bf*)(bp + 16);
        v16bf bfrag = __builtin_shufflevector(blo, bhi,
            0,1,2,3,4,5,6,7,8,9,10,11,12,13,14,15);
        acc[j] = __builtin_amdgcn_wmma_f32_16x16x32_bf16(
            false, afrag, false, bfrag, (short)0, acc[j], false, false);
      }
    }
    __syncthreads();   // readers done before buffers are overwritten

    ky = ky2; kx = kx2; cc = cc2;
  }

  // C/D layout: lane L holds N = L&15, rows M = v + 8*(L>>4) in VGPR v
  const int mrow0 = mBase + waveM * 16 + ((lane >> 4) << 3);
#pragma unroll
  for (int j = 0; j < 4; ++j) {
    const int co = nBase + waveN * 64 + j * 16 + l16;
    if (co >= Cout) continue;
    const float sc = scale ? scale[co] : 1.0f;
    const float bi = bias ? bias[co] : 0.0f;
#pragma unroll
    for (int v = 0; v < 8; ++v) {
      const int mm = mrow0 + v;
      if (mm >= M) continue;
      float val = acc[j][v] * sc + bi;
      if (res)  val += (float)res[(size_t)mm * Cout + co];
      if (relu) val = fmaxf(val, 0.0f);
      out[(size_t)mm * Cout + co] = (bf16)val;
    }
  }
}

// OIHW fp32 -> [Cout][(ky*kw+kx)*Cin+ci] bf16
__global__ void __launch_bounds__(256)
repack_w(const float* __restrict__ Wsrc, bf16* __restrict__ Wp,
         int Cout, int Cin, int kh, int kw)
{
  const long K = (long)Cin * kh * kw;
  const long total = (long)Cout * K;
  const long i = (long)blockIdx.x * 256 + threadIdx.x;
  if (i >= total) return;
  const int co = (int)(i / K);
  const int k  = (int)(i - (long)co * K);
  const int ci = k % Cin;
  const int f  = k / Cin;
  const int ky = f / kw;
  const int kx = f - ky * kw;
  Wp[i] = (bf16)Wsrc[(((size_t)co * Cin + ci) * kh + ky) * kw + kx];
}

// Stem: 7x7 s2 p3 conv on (32,3,256,128) NCHW fp32 -> (32,128,64,64) NHWC bf16,
// fused BN + ReLU. Cin=3 is too small for WMMA; VALU direct conv.
__global__ void __launch_bounds__(256)
stem_conv(const float* __restrict__ x, const float* __restrict__ w,
          const float* __restrict__ s, const float* __restrict__ b,
          bf16* __restrict__ out)
{
  const int idx = blockIdx.x * 256 + threadIdx.x;
  const int total = 32 * 128 * 64 * 64;
  if (idx >= total) return;
  const int co = idx & 63;
  int t = idx >> 6;
  const int wo = t % 64;  t /= 64;
  const int ho = t % 128;
  const int nb = t / 128;
  float acc = 0.0f;
  for (int ci = 0; ci < 3; ++ci)
    for (int ky = 0; ky < 7; ++ky) {
      const int hi = ho * 2 - 3 + ky;
      if (hi < 0 || hi >= 256) continue;
      for (int kx = 0; kx < 7; ++kx) {
        const int wi = wo * 2 - 3 + kx;
        if (wi < 0 || wi >= 128) continue;
        acc += x[((size_t)(nb * 3 + ci) * 256 + hi) * 128 + wi] *
               w[((size_t)(co * 3 + ci) * 7 + ky) * 7 + kx];
      }
    }
  acc = acc * s[co] + b[co];
  out[idx] = (bf16)fmaxf(acc, 0.0f);
}

// 3x3 s2 p1 maxpool: (32,128,64,64) NHWC -> (32,64,32,64) NHWC
__global__ void __launch_bounds__(256)
maxpool3x3(const bf16* __restrict__ in, bf16* __restrict__ out)
{
  const int idx = blockIdx.x * 256 + threadIdx.x;
  const int total = 32 * 64 * 32 * 64;
  if (idx >= total) return;
  const int co = idx & 63;
  int t = idx >> 6;
  const int wo = t % 32;  t /= 32;
  const int ho = t % 64;
  const int nb = t / 64;
  float m = -3.0e38f;
  for (int ky = 0; ky < 3; ++ky) {
    const int hi = ho * 2 - 1 + ky;
    if (hi < 0 || hi >= 128) continue;
    for (int kx = 0; kx < 3; ++kx) {
      const int wi = wo * 2 - 1 + kx;
      if (wi < 0 || wi >= 64) continue;
      m = fmaxf(m, (float)in[(((size_t)nb * 128 + hi) * 64 + wi) * 64 + co]);
    }
  }
  out[idx] = (bf16)m;
}

// Global average pool: (32,8,4,2048) NHWC -> (32,2048) bf16
__global__ void __launch_bounds__(256)
avgpool(const bf16* __restrict__ in, bf16* __restrict__ out)
{
  const int idx = blockIdx.x * 256 + threadIdx.x;
  if (idx >= 32 * 2048) return;
  const int c  = idx & 2047;
  const int nb = idx >> 11;
  float s = 0.0f;
  for (int p = 0; p < 32; ++p)
    s += (float)in[((size_t)nb * 32 + p) * 2048 + c];
  out[idx] = (bf16)(s * (1.0f / 32.0f));
}

// 6-qubit, 16-layer RX+CNOT simulation. One wave32 per (batch, chunk):
// 64 complex amplitudes -> 2 per lane (index = k*32 + lane, k = bit 5).
// Bits 0..4 pair via __shfl_xor; bit 5 pairs in-lane across the k registers.
__global__ void __launch_bounds__(256)
quantum_kernel(const bf16* __restrict__ h, const float* __restrict__ qw,
               float* __restrict__ q)
{
  const int wave = (blockIdx.x * 256 + threadIdx.x) >> 5;
  const int lane = threadIdx.x & 31;
  if (wave >= 512) return;
  const int b = wave >> 4;
  const int c = wave & 15;

  float a0 = (float)h[(size_t)wave * 64 + lane];
  float a1 = (float)h[(size_t)wave * 64 + 32 + lane];
  const float nrm = wave_sum(a0 * a0 + a1 * a1);
  const float inv = rsqrtf(fmaxf(nrm, 1e-30f));
  float re0 = a0 * inv, im0 = 0.0f;
  float re1 = a1 * inv, im1 = 0.0f;

  for (int l = 0; l < 16; ++l) {
    for (int w = 0; w < 6; ++w) {
      const float th = qw[((size_t)c * 16 + l) * 6 + w];
      const float ct = cosf(0.5f * th);
      const float st = sinf(0.5f * th);
      float pr0, pi0, pr1, pi1;
      if (w == 0) {                       // bit 5: partner is other k register
        pr0 = re1; pi0 = im1; pr1 = re0; pi1 = im0;
      } else {
        const int msk = 1 << (5 - w);
        pr0 = __shfl_xor(re0, msk, 32); pi0 = __shfl_xor(im0, msk, 32);
        pr1 = __shfl_xor(re1, msk, 32); pi1 = __shfl_xor(im1, msk, 32);
      }
      const float nr0 = ct * re0 + st * pi0;
      const float ni0 = ct * im0 - st * pr0;
      const float nr1 = ct * re1 + st * pi1;
      const float ni1 = ct * im1 - st * pr1;
      re0 = nr0; im0 = ni0; re1 = nr1; im1 = ni1;
    }
    for (int w = 0; w < 6; ++w) {
      const int cb = 5 - w;
      const int tb = (w == 5) ? 5 : (4 - w);
      float fr0, fi0, fr1, fi1;
      if (tb == 5) {
        fr0 = re1; fi0 = im1; fr1 = re0; fi1 = im0;
      } else {
        const int msk = 1 << tb;
        fr0 = __shfl_xor(re0, msk, 32); fi0 = __shfl_xor(im0, msk, 32);
        fr1 = __shfl_xor(re1, msk, 32); fi1 = __shfl_xor(im1, msk, 32);
      }
      const int c0 = (cb == 5) ? 0 : ((lane >> cb) & 1);
      const int c1 = (cb == 5) ? 1 : ((lane >> cb) & 1);
      if (c0) { re0 = fr0; im0 = fi0; }
      if (c1) { re1 = fr1; im1 = fi1; }
    }
  }

  const float p0 = re0 * re0 + im0 * im0;
  const float p1 = re1 * re1 + im1 * im1;
#pragma unroll
  for (int w = 0; w < 6; ++w) {
    const int bit = 5 - w;
    const float s0 = 1.0f - 2.0f * (float)((lane >> bit) & 1);
    const float s1 = 1.0f - 2.0f * (float)(((32 + lane) >> bit) & 1);
    const float e  = wave_sum(s0 * p0 + s1 * p1);
    if (lane == 0) q[(size_t)b * 96 + c * 6 + w] = e;
  }
}

// fc2: (32,96) fp32 x (200,96)^T + bias -> (32,200) fp32
__global__ void __launch_bounds__(256)
fc2_kernel(const float* __restrict__ q, const float* __restrict__ w,
           const float* __restrict__ bias, float* __restrict__ out)
{
  const int idx = blockIdx.x * 256 + threadIdx.x;
  if (idx >= 32 * 200) return;
  const int o = idx % 200;
  const int b = idx / 200;
  float s = bias[o];
  for (int k = 0; k < 96; ++k) s += q[b * 96 + k] * w[o * 96 + k];
  out[idx] = s;
}

// ---------------------------------------------------------------------------
// Host side
// ---------------------------------------------------------------------------
static void launch_repack(hipStream_t st, const float* w, bf16* wp,
                          int Cout, int Cin, int kh, int kw) {
  long total = (long)Cout * Cin * kh * kw;
  int blocks = (int)((total + 255) / 256);
  repack_w<<<blocks, 256, 0, st>>>(w, wp, Cout, Cin, kh, kw);
}

static void launch_gemm(hipStream_t st, const bf16* A, const bf16* Wp,
                        const float* scale, const float* bias, const bf16* res,
                        bf16* out, int Nb, int H, int W, int Cin, int Cout,
                        int kh, int kw, int stride, int pad, int Ho, int Wo, int relu) {
  int M = Nb * Ho * Wo;
  dim3 grid((Cout + 127) / 128, (M + 63) / 64);
  gemm_conv_wmma<<<grid, 256, 0, st>>>(A, Wp, scale, bias, res, out,
                                       Nb, H, W, Cin, Cout, kh, kw, stride, pad,
                                       Ho, Wo, relu);
}

extern "C" void kernel_launch(void* const* d_in, const int* in_sizes, int n_in,
                              void* d_out, int out_size, void* d_ws, size_t ws_size,
                              hipStream_t stream)
{
  (void)in_sizes; (void)out_size; (void)ws_size;
  if (n_in < 165) return;   // unexpected flattening; deterministic no-op

  // Block configuration (mirrors reference _block_cfg)
  int cfg_cin[16], cfg_mid[16], cfg_cout[16], cfg_st[16], cfg_dn[16];
  {
    const int mids[4]   = {64, 128, 256, 512};
    const int counts[4] = {3, 4, 6, 3};
    const int strides[4]= {1, 2, 2, 2};
    int cin = 64, n = 0;
    for (int g = 0; g < 4; ++g) {
      const int mid = mids[g], cout = mid * 4;
      for (int j = 0; j < counts[g]; ++j) {
        cfg_cin[n] = cin; cfg_mid[n] = mid; cfg_cout[n] = cout;
        cfg_st[n] = (j == 0 ? strides[g] : 1); cfg_dn[n] = (j == 0);
        cin = cout; ++n;
      }
    }
  }

  // Input walk in setup_inputs insertion order
  int ix = 0;
  const float* X      = (const float*)d_in[ix++];
  const float* stem_w = (const float*)d_in[ix++];
  const float* stem_s = (const float*)d_in[ix++];
  const float* stem_b = (const float*)d_in[ix++];
  struct BP { const float *w1,*w2,*w3,*s1,*b1,*s2,*b2,*s3,*b3,*wd,*sd,*bd; };
  BP bp[16];
  for (int i = 0; i < 16; ++i) {
    bp[i].w1 = (const float*)d_in[ix++];
    bp[i].w2 = (const float*)d_in[ix++];
    bp[i].w3 = (const float*)d_in[ix++];
    bp[i].s1 = (const float*)d_in[ix++];
    bp[i].b1 = (const float*)d_in[ix++];
    bp[i].s2 = (const float*)d_in[ix++];
    bp[i].b2 = (const float*)d_in[ix++];
    bp[i].s3 = (const float*)d_in[ix++];
    bp[i].b3 = (const float*)d_in[ix++];
    if (cfg_dn[i]) {
      bp[i].wd = (const float*)d_in[ix++];
      bp[i].sd = (const float*)d_in[ix++];
      bp[i].bd = (const float*)d_in[ix++];
    } else { bp[i].wd = nullptr; bp[i].sd = nullptr; bp[i].bd = nullptr; }
  }
  const float* fc1_w = (const float*)d_in[ix++];
  const float* fc1_b = (const float*)d_in[ix++];
  const float* qwts  = (const float*)d_in[ix++];
  const float* fc2_w = (const float*)d_in[ix++];
  const float* fc2_b = (const float*)d_in[ix++];

  // Workspace layout
  char* base = (char*)d_ws;
  const size_t MB = 1024ull * 1024ull;
  bf16* T0  = (bf16*)(base +   0 * MB);   // x (post-maxpool / block inputs)
  bf16* T1  = (bf16*)(base +  34 * MB);   // conv1 out / stem out
  bf16* T2  = (bf16*)(base +  68 * MB);   // conv2 out
  bf16* T3  = (bf16*)(base + 102 * MB);   // conv3 out (ping-pong with T0)
  bf16* T4  = (bf16*)(base + 136 * MB);   // downsample identity
  bf16* WPK = (bf16*)(base + 170 * MB);   // repacked weights (<= 4.7 MB)
  bf16* FE  = (bf16*)(base + 179 * MB);   // feats (32,2048)
  bf16* HH  = (bf16*)(base + 180 * MB);   // fc1 out (32,1024)
  float* QV = (float*)(base + 181 * MB);  // quantum out (32,96)

  // Stem + maxpool
  stem_conv<<<(32*128*64*64) / 256, 256, 0, stream>>>(X, stem_w, stem_s, stem_b, T1);
  maxpool3x3<<<(32*64*32*64 + 255) / 256, 256, 0, stream>>>(T1, T0);

  // Bottleneck blocks
  bf16* cur = T0;
  bf16* spare = T3;
  int H = 64, W = 32;
  for (int i = 0; i < 16; ++i) {
    const int Cin = cfg_cin[i], mid = cfg_mid[i], Cout = cfg_cout[i], st = cfg_st[i];
    const int Ho = (H + 2 - 3) / st + 1;
    const int Wo = (W + 2 - 3) / st + 1;

    launch_repack(stream, bp[i].w1, WPK, mid, Cin, 1, 1);
    launch_gemm(stream, cur, WPK, bp[i].s1, bp[i].b1, nullptr, T1,
                32, H, W, Cin, mid, 1, 1, 1, 0, H, W, 1);

    launch_repack(stream, bp[i].w2, WPK, mid, mid, 3, 3);
    launch_gemm(stream, T1, WPK, bp[i].s2, bp[i].b2, nullptr, T2,
                32, H, W, mid, mid, 3, 3, st, 1, Ho, Wo, 1);

    const bf16* resptr;
    if (cfg_dn[i]) {
      launch_repack(stream, bp[i].wd, WPK, Cout, Cin, 1, 1);
      launch_gemm(stream, cur, WPK, bp[i].sd, bp[i].bd, nullptr, T4,
                  32, H, W, Cin, Cout, 1, 1, st, 0, Ho, Wo, 0);
      resptr = T4;
    } else {
      resptr = cur;
    }

    launch_repack(stream, bp[i].w3, WPK, Cout, mid, 1, 1);
    launch_gemm(stream, T2, WPK, bp[i].s3, bp[i].b3, resptr, spare,
                32, Ho, Wo, mid, Cout, 1, 1, 1, 0, Ho, Wo, 1);

    bf16* t = cur; cur = spare; spare = t;
    H = Ho; W = Wo;
  }

  // Head: avgpool -> fc1 (WMMA GEMM) -> quantum -> fc2
  avgpool<<<(32*2048 + 255) / 256, 256, 0, stream>>>(cur, FE);

  launch_repack(stream, fc1_w, WPK, 1024, 2048, 1, 1);
  launch_gemm(stream, FE, WPK, nullptr, fc1_b, nullptr, HH,
              32, 1, 1, 2048, 1024, 1, 1, 1, 0, 1, 1, 0);

  quantum_kernel<<<64, 256, 0, stream>>>(HH, qwts, QV);
  fc2_kernel<<<(32*200 + 255) / 256, 256, 0, stream>>>(QV, fc2_w, fc2_b, (float*)d_out);
}
